// GRU_2181843386434
// MI455X (gfx1250) — matrix operand order repaired and checked by reading, not compiled, not run on hardware
//
#include <hip/hip_runtime.h>
#include <stdint.h>

#define BATCH  256
#define SEQ    512
#define INPUT  128
#define HIDDEN 256
#define KTOT   (HIDDEN + INPUT)   // 384 = concat(h, x) reduction dim

typedef __attribute__((ext_vector_type(16))) __bf16 v16bf;
typedef __attribute__((ext_vector_type(8)))  float  v8f;

union ABf {
  v16bf v;
  uint4 q[2];
};

__device__ __forceinline__ unsigned short f2bf(float f) {
  unsigned int u = __float_as_uint(f);
  unsigned int r = u + 0x7FFFu + ((u >> 16) & 1u);   // round-to-nearest-even
  return (unsigned short)(r >> 16);
}
__device__ __forceinline__ float bf2f(unsigned short s) {
  return __uint_as_float(((unsigned int)s) << 16);
}
__device__ __forceinline__ float fsigmoid(float x) { return 1.0f / (1.0f + __expf(-x)); }
__device__ __forceinline__ float ftanh(float x) {
  float e = __expf(2.0f * x);
  return (e - 1.0f) / (e + 1.0f);
}

// LDS row strides (elements), padded so 16 rows map to distinct banks
#define SH 264   // for 256-wide h / r*h buffers (528B pitch, 16B aligned)
#define SX 136   // for 128-wide x buffer      (272B pitch, 16B aligned)

// A operand (16x32 bf16): lane holds row M=lane%16; lanes<16 -> K {0..7,16..23},
// lanes>=16 -> K {8..15,24..31} (ISA 7.12.2 16-bit A layout). Two ds_load_b128.
__device__ __forceinline__ void load_a_lds(const unsigned short* base, int stride,
                                           int col0, int lane, ABf& a) {
  int m    = lane & 15;
  int koff = (lane >> 4) << 3;          // 0 or 8
  const unsigned short* p = base + m * stride + col0 + koff;
  a.q[0] = *(const uint4*)(p);
  a.q[1] = *(const uint4*)(p + 16);
}

// B operand (32x16 bf16): lane holds column N=lane%16; lanes<16 -> K {0..7,16..23},
// lanes>=16 -> K {8..15,24..31}. Per-lane two 16-byte chunks along W's input dim:
// two global_load_b128 from the packed bf16 weights (L2-resident).
__device__ __forceinline__ void load_b_w(const unsigned short* Wrow, int k0, ABf& b) {
  const unsigned short* p = Wrow + k0;
  b.q[0] = *(const uint4*)(p);
  b.q[1] = *(const uint4*)(p + 16);
}

// Phase 0: convert W_z/W_r/W_c f32 [256,384] -> bf16 packed [3][256][384] in ws.
__global__ void pack_weights_kernel(const float* __restrict__ Wz,
                                    const float* __restrict__ Wr,
                                    const float* __restrict__ Wc,
                                    unsigned short* __restrict__ out) {
  const int NW = HIDDEN * KTOT;
  int idx = blockIdx.x * blockDim.x + threadIdx.x;
  if (idx >= 3 * NW) return;
  const float* src = (idx < NW) ? Wz : ((idx < 2 * NW) ? Wr : Wc);
  out[idx] = f2bf(src[idx % NW]);
}

// Persistent recurrent kernel: 16 blocks x 512 threads (16 waves).
// Block owns batch rows [bt, bt+16); wave w owns hidden cols [16w, 16w+16).
__global__ __launch_bounds__(512) void gru_seq_kernel(
    const float* __restrict__ X,
    const unsigned short* __restrict__ Wpack,   // bf16 [3][256][384]: z, r, c
    const float* __restrict__ b_z, const float* __restrict__ b_r,
    const float* __restrict__ b_c,
    float* __restrict__ out) {
  __shared__ __align__(16) unsigned short hbuf[2][16][SH];  // h, double buffered
  __shared__ __align__(16) unsigned short rhbuf[16][SH];    // r*h
  __shared__ __align__(16) unsigned short xbuf[16][SX];     // x_t (bf16)

  const int tid  = threadIdx.x;
  const int lane = tid & 31;
  const int wid  = tid >> 5;            // 0..15
  const int n0   = wid * 16;
  const int bt   = blockIdx.x * 16;

  // Per-lane element offset into packed W for B tiles:
  // row = n0 + lane%16, +8 K-columns for the hi half-wave.
  const int browoff = (n0 + (lane & 15)) * KTOT + ((lane >> 4) << 3);

  // C/D f32 tile mapping: (vgpr i, lane) -> M = m_base + i, N = ncol
  const int m_base = (lane >> 4) << 3;  // 0 or 8
  const int ncol   = n0 + (lane & 15);

  const float bzs = b_z[ncol];
  const float brs = b_r[ncol];
  const float bcs = b_c[ncol];

  // h0 = 0
#pragma unroll
  for (int i = 0; i < 8; ++i) hbuf[0][m_base + i][ncol] = 0;
  __syncthreads();

  const int xm = tid >> 5;              // x staging: row 0..15
  const int xc = (tid & 31) * 4;        // 4 cols per thread

  // Opaque (always-zero) uniform offset, redefined by volatile asm every
  // iteration: defeats LICM (which would hoist 36 B tiles = 288 VGPRs and
  // spill to scratch) while keeping Wpack's global address space inference
  // intact so the weight streams stay global_load_b128 from L2.
  unsigned int wofs = 0;

  int p = 0;
  for (int t = 0; t < SEQ; ++t) {
    asm volatile("" : "+s"(wofs));      // uniform: SGPR constraint is legal
    const unsigned short* Wzr = Wpack + wofs + browoff;
    const unsigned short* Wrr = Wzr + HIDDEN * KTOT;
    const unsigned short* Wcr = Wrr + HIDDEN * KTOT;

    // Stage x[bt+xm, t, xc..xc+3] -> bf16 LDS tile
    {
      const float4 xv =
          *(const float4*)(X + (((size_t)(bt + xm) * SEQ + t) * INPUT + xc));
      unsigned int lo = (unsigned int)f2bf(xv.x) | ((unsigned int)f2bf(xv.y) << 16);
      unsigned int hi = (unsigned int)f2bf(xv.z) | ((unsigned int)f2bf(xv.w) << 16);
      *(uint2*)(&xbuf[xm][xc]) = make_uint2(lo, hi);
    }
    __syncthreads();

    // ---- Gate GEMMs: z, r over K = [h(256) | x(128)], shared A operand ----
    v8f zacc = {};
    v8f racc = {};
#pragma unroll
    for (int kt = 0; kt < 12; ++kt) {
      ABf a;
      if (kt < 8) load_a_lds(&hbuf[p][0][0], SH, kt * 32, lane, a);
      else        load_a_lds(&xbuf[0][0],    SX, (kt - 8) * 32, lane, a);
      ABf bz_t, br_t;
      load_b_w(Wzr, kt * 32, bz_t);
      load_b_w(Wrr, kt * 32, br_t);
      zacc = __builtin_amdgcn_wmma_f32_16x16x32_bf16(
          false, a.v, false, bz_t.v, (short)0, zacc, false, false);
      racc = __builtin_amdgcn_wmma_f32_16x16x32_bf16(
          false, a.v, false, br_t.v, (short)0, racc, false, false);
    }

    float zv[8], hold[8];
#pragma unroll
    for (int i = 0; i < 8; ++i) {
      int m   = m_base + i;
      zv[i]   = fsigmoid(zacc[i] + bzs);
      float r = fsigmoid(racc[i] + brs);
      hold[i] = bf2f(hbuf[p][m][ncol]);
      rhbuf[m][ncol] = f2bf(r * hold[i]);
    }
    __syncthreads();

    // ---- Candidate GEMM over K = [r*h(256) | x(128)] ----
    v8f cacc = {};
#pragma unroll
    for (int kt = 0; kt < 12; ++kt) {
      ABf a;
      if (kt < 8) load_a_lds(&rhbuf[0][0], SH, kt * 32, lane, a);
      else        load_a_lds(&xbuf[0][0],  SX, (kt - 8) * 32, lane, a);
      ABf bc_t;
      load_b_w(Wcr, kt * 32, bc_t);
      cacc = __builtin_amdgcn_wmma_f32_16x16x32_bf16(
          false, a.v, false, bc_t.v, (short)0, cacc, false, false);
    }

#pragma unroll
    for (int i = 0; i < 8; ++i) {
      int m    = m_base + i;
      float cv = ftanh(cacc[i] + bcs);
      float hn = (1.0f - zv[i]) * hold[i] + zv[i] * cv;
      out[((size_t)t * BATCH + (bt + m)) * HIDDEN + ncol] = hn;  // [seq,batch,hidden]
      hbuf[p ^ 1][m][ncol] = f2bf(hn);
    }
    __syncthreads();
    p ^= 1;
  }
}

extern "C" void kernel_launch(void* const* d_in, const int* in_sizes, int n_in,
                              void* d_out, int out_size, void* d_ws, size_t ws_size,
                              hipStream_t stream) {
  const float* X  = (const float*)d_in[0];
  const float* Wz = (const float*)d_in[1];
  const float* bz = (const float*)d_in[2];
  const float* Wr = (const float*)d_in[3];
  const float* br = (const float*)d_in[4];
  const float* Wc = (const float*)d_in[5];
  const float* bc = (const float*)d_in[6];

  unsigned short* Wpack = (unsigned short*)d_ws;   // 3*256*384*2 = 576 KB
  const int NW = HIDDEN * KTOT;

  pack_weights_kernel<<<(3 * NW + 255) / 256, 256, 0, stream>>>(Wz, Wr, Wc, Wpack);
  gru_seq_kernel<<<BATCH / 16, 512, 0, stream>>>(X, Wpack, bz, br, bc,
                                                 (float*)d_out);
}
